// painn_25778393710979
// MI455X (gfx1250) — matrix operand order repaired and checked by reading, not compiled, not run on hardware
//
#include <hip/hip_runtime.h>
#include <math.h>

// ---------------------------------------------------------------------------
// PaiNN-style forward for MI455X (gfx1250, wave32).
// - All matmuls: v_wmma_f32_16x16x32_bf16, f32 accumulate, K fixed at 128.
// - A row-tiles staged into LDS via Tensor Data Mover (tensor_load_to_lds),
//   synchronized with s_wait_tensorcnt + workgroup barrier.
// - GEMM epilogues fuse bias (+ sqrt(scal)*bias2), SiLU, bf16 requantization.
// ---------------------------------------------------------------------------

typedef __attribute__((ext_vector_type(16))) __bf16 bf16x16;
typedef __attribute__((ext_vector_type(8)))  __bf16 bf16x8;
typedef __attribute__((ext_vector_type(8)))  float  f32x8;
typedef __attribute__((ext_vector_type(4)))  unsigned int uint32x4;
typedef __attribute__((ext_vector_type(8)))  int    int32x8;
typedef __attribute__((ext_vector_type(4)))  int    int32x4;

#define DD   128
#define TD   384
#define GK   128          // contraction dim of every GEMM in this network
#define PI_F 3.14159265358979f

__device__ __forceinline__ __bf16 f2b(float f) {
    union { float f; unsigned u; } x; x.f = f;
    unsigned r = (x.u + 0x7FFFu + ((x.u >> 16) & 1u)) >> 16;
    union { unsigned short s; __bf16 b; } y;
    y.s = (unsigned short)r;
    return y.b;
}

// ---------------- utility kernels ----------------

__global__ void zero_kernel(float* p, size_t n) {
    size_t i  = (size_t)blockIdx.x * blockDim.x + threadIdx.x;
    size_t st = (size_t)gridDim.x * blockDim.x;
    for (; i < n; i += st) p[i] = 0.f;
}

// W is (K x Nout) row-major f32 (optionally starting at row `rowoff`),
// WT is (Nout x K) row-major bf16.
__global__ void wt_kernel(const float* W, __bf16* WT, int K, int Nout, int rowoff) {
    int idx = blockIdx.x * 256 + threadIdx.x;
    if (idx >= Nout * K) return;
    int k = idx % K;
    int n = idx / K;
    WT[idx] = f2b(W[(size_t)(k + rowoff) * Nout + n]);
}

// c1[n] = sum_{k<128} Wa1[k][n]   (colsum of upper half of Wa1, 256x128)
__global__ void c1_kernel(const float* Wa1, float* c1) {
    int n = threadIdx.x;
    float s = 0.f;
    for (int k = 0; k < DD; ++k) s += Wa1[(size_t)k * DD + n];
    c1[n] = s;
}

__global__ void embed_kernel(const int* atomic, const float* emb,
                             float* S32, __bf16* S16, size_t n) {
    size_t idx = (size_t)blockIdx.x * 256 + threadIdx.x;
    if (idx >= n) return;
    int    c = (int)(idx & (DD - 1));
    size_t i = idx >> 7;
    int    t = atomic[i];
    float  s = emb[(size_t)t * DD + c];
    S32[idx] = s;
    S16[idx] = f2b(s);
}

// rn[i] = |r_i| ; frobacc += x^2+y^2+z^2   (block tree-reduce then 1 atomic)
__global__ void rn_kernel(const float* r, float* rn, float* frobacc, int N) {
    __shared__ float ls[256];
    int i = blockIdx.x * 256 + threadIdx.x;
    float p = 0.f;
    if (i < N) {
        float x = r[(size_t)i * 3 + 0];
        float y = r[(size_t)i * 3 + 1];
        float z = r[(size_t)i * 3 + 2];
        float q = x * x + y * y + z * z;
        rn[i] = sqrtf(q);
        p = q;
    }
    ls[threadIdx.x] = p;
    __syncthreads();
    for (int o = 128; o > 0; o >>= 1) {
        if (threadIdx.x < o) ls[threadIdx.x] += ls[threadIdx.x + o];
        __syncthreads();
    }
    if (threadIdx.x == 0) atomicAdd(frobacc, ls[0]);
}

__global__ void k_rsqrt(float* p) { p[0] = 1.0f / sqrtf(p[0]); }

__global__ void reduce_sq(const float* x, size_t n, float* acc) {
    __shared__ float ls[256];
    size_t i  = (size_t)blockIdx.x * 256 + threadIdx.x;
    size_t st = (size_t)gridDim.x * 256;
    float s = 0.f;
    for (; i < n; i += st) { float v = x[i]; s += v * v; }
    ls[threadIdx.x] = s; __syncthreads();
    for (int o = 128; o > 0; o >>= 1) {
        if (threadIdx.x < o) ls[threadIdx.x] += ls[threadIdx.x + o];
        __syncthreads();
    }
    if (threadIdx.x == 0) atomicAdd(acc, ls[0]);
}

__global__ void reduce_dot(const float* x, const float* y, size_t n, float* acc) {
    __shared__ float ls[256];
    size_t i  = (size_t)blockIdx.x * 256 + threadIdx.x;
    size_t st = (size_t)gridDim.x * 256;
    float s = 0.f;
    for (; i < n; i += st) s += x[i] * y[i];
    ls[threadIdx.x] = s; __syncthreads();
    for (int o = 128; o > 0; o >>= 1) {
        if (threadIdx.x < o) ls[threadIdx.x] += ls[threadIdx.x + o];
        __syncthreads();
    }
    if (threadIdx.x == 0) atomicAdd(acc, ls[0]);
}

__global__ void reduce_sum(const float* x, size_t n, float* acc) {
    __shared__ float ls[256];
    size_t i  = (size_t)blockIdx.x * 256 + threadIdx.x;
    size_t st = (size_t)gridDim.x * 256;
    float s = 0.f;
    for (; i < n; i += st) s += x[i];
    ls[threadIdx.x] = s; __syncthreads();
    for (int o = 128; o > 0; o >>= 1) {
        if (threadIdx.x < o) ls[threadIdx.x] += ls[threadIdx.x + o];
        __syncthreads();
    }
    if (threadIdx.x == 0) atomicAdd(acc, ls[0]);
}

// ---------------- WMMA bf16 GEMM, TDM-staged A, fused epilogue ----------------
// C[M x Nc] = act( A[M x GK] * B[GK x Nc] + bias (+ sqrt?(scal)*bias2) )
// A: bf16 row-major.  BT: bf16 row-major (Nc x GK) == B transposed.
// One block (4 waves) owns a 16-row tile:
//   * each wave TDM-loads 4 rows of A into LDS (tensor_load_to_lds),
//   * A K-fragments hoisted to registers once (4 x bf16x16),
//   * waves loop over 16-col tiles, reloading only B (cache-resident weights).
// Requires M % 16 == 0 (true for all call sites here).
__global__ void gemm_wmma(const __bf16* __restrict__ A,
                          const __bf16* __restrict__ BT,
                          const float* __restrict__ bias,
                          const float* __restrict__ bias2,
                          const float* __restrict__ scal, int scal_sqrt,
                          float* __restrict__ C32, __bf16* __restrict__ C16,
                          long M, int Nc, int act) {
    __shared__ __align__(16) __bf16 lA[16 * GK];     // 4 KB
    int wave = threadIdx.x >> 5;
    int lane = threadIdx.x & 31;
    long row0 = (long)blockIdx.x << 4;

    // ---- Tensor Data Mover: stage 4 rows of A per wave into LDS ----
    {
        size_t   ga   = (size_t)A + ((size_t)(row0 + 4 * wave) * GK) * 2;
        unsigned lbas = (unsigned)(size_t)(&lA[0]) + (unsigned)(4 * wave * GK * 2);
        uint32x4 g0;
        g0.x = 1u;                                        // count=1, user mode
        g0.y = lbas;                                      // lds_addr (bytes)
        g0.z = (unsigned)(ga & 0xFFFFFFFFu);              // global_addr[31:0]
        g0.w = (unsigned)((ga >> 32) & 0x01FFFFFFu)       // global_addr[56:32]
             | (2u << 30);                                // type = 2 ("image")
        int32x8 g1;
        g1[0] = (1 << 16);              // data_size = 2 bytes
        g1[1] = (GK << 16);             // tensor_dim0 = GK      (bits 63:48)
        g1[2] = (4 << 16);              // tensor_dim1 = 4       (bits 95:80)
        g1[3] = (GK << 16);             // tile_dim0   = GK      (bits 127:112)
        g1[4] = 4;                      // tile_dim1   = 4, tile_dim2 = 0
        g1[5] = GK;                     // tensor_dim0_stride = GK
        g1[6] = 0;
        g1[7] = 0;
        int32x4 gz4 = {0, 0, 0, 0};
        int32x8 gz8 = {0, 0, 0, 0, 0, 0, 0, 0};
        __builtin_amdgcn_tensor_load_to_lds(g0, g1, gz4, gz4, gz8, 0);
        __builtin_amdgcn_s_wait_tensorcnt(0);
    }
    __syncthreads();

    int  l16 = lane & 15;
    bool hi  = lane >= 16;

    // A fragments from LDS (ds_load_b128), hoisted across all column tiles.
    // 16-bit A layout: lanes 0-15 -> K k0..k0+7 | k0+16..k0+23,
    //                  lanes 16-31 -> K k0+8..k0+15 | k0+24..k0+31.
    const __bf16* pA = lA + l16 * GK + (hi ? 8 : 0);
    bf16x16 af[GK / 32];
    {
        union { bf16x16 v; bf16x8 h[2]; } t;
        #pragma unroll
        for (int kk = 0; kk < GK / 32; ++kk) {
            t.h[0] = *(const bf16x8*)(pA + kk * 32);
            t.h[1] = *(const bf16x8*)(pA + kk * 32 + 16);
            af[kk] = t.v;
        }
    }

    int nTilesN = Nc >> 4;
    for (int tileN = wave; tileN < nTilesN; tileN += 4) {
        int col0 = tileN << 4;
        // B (32x16 bf16/step): lanes 0-15 hold K k0..k0+7 | k0+16..k0+23 of column l16,
        //                      lanes 16-31 hold K k0+8..k0+15 | k0+24..k0+31.
        const __bf16* pB = BT + (size_t)(col0 + l16) * GK + (hi ? 8 : 0);
        __builtin_prefetch(pB, 0, 1);
        f32x8 c = {};
        #pragma unroll
        for (int kk = 0; kk < GK / 32; ++kk) {
            union { bf16x16 v; bf16x8 h[2]; } tb;
            tb.h[0] = *(const bf16x8*)(pB + kk * 32);
            tb.h[1] = *(const bf16x8*)(pB + kk * 32 + 16);
            bf16x16 b = tb.v;
            c = __builtin_amdgcn_wmma_f32_16x16x32_bf16(
                    false, af[kk], false, b, (short)0, c, false, false);
        }

        int col = col0 + l16;
        float be = bias ? bias[col] : 0.f;
        if (bias2) {
            float sc = scal[0];
            if (scal_sqrt) sc = sqrtf(sc);
            be += sc * bias2[col];
        }
        long rbase = row0 + (hi ? 8 : 0);
        #pragma unroll
        for (int i = 0; i < 8; ++i) {
            float x = c[i] + be;
            if (act) x = x / (1.f + __expf(-x));   // SiLU
            size_t o = (size_t)(rbase + i) * Nc + col;
            if (C32) C32[o] = x;
            if (C16) C16[o] = f2b(x);
        }
    }
}

// ---------------- message: RBF -> cutoff -> gate -> column reductions -------
// OSV[0..127]   += sum_i g1[i][c]
// OSV[128..511] += sum_i ( g0[i][c]*v[i][d][c] + g2[i][c]*rn_i/frob )
__global__ void gate_reduce_kernel(const float* __restrict__ phi,
                                   const float* __restrict__ V32,
                                   const float* __restrict__ rn,
                                   const float* __restrict__ invfrob,
                                   const float* __restrict__ Wr,
                                   const float* __restrict__ br,
                                   float* __restrict__ OSV, int N) {
    __shared__ float ls[512];
    int t = threadIdx.x;                        // 0..127, 4 waves
    for (int k = t; k < 512; k += 128) ls[k] = 0.f;
    __syncthreads();

    int wave = t >> 5, lane = t & 31;
    int gw = blockIdx.x * 4 + wave;
    int nw = gridDim.x * 4;
    float accS[4] = {0.f, 0.f, 0.f, 0.f};
    float accV[4][3] = {};
    float invf = invfrob[0];

    for (int i = gw; i < N; i += nw) {
        float rni    = rn[i];
        float orgn   = rni * invf;
        float inv_rn = 1.f / rni;
        float base   = rni * (PI_F * 0.5f);     // pi/R_CUT, R_CUT=2
        float sv[20];
        #pragma unroll
        for (int j = 0; j < 20; ++j)
            sv[j] = __sinf((float)(j + 1) * base) * inv_rn;

        const float* ph = phi + (size_t)i * TD;
        const float* vp = V32 + (size_t)i * TD;
        #pragma unroll
        for (int q = 0; q < 4; ++q) {
            int c = lane + 32 * q;
            float g[3];
            #pragma unroll
            for (int gi = 0; gi < 3; ++gi) {
                int col = gi * DD + c;
                float rp = br[col];
                #pragma unroll
                for (int j = 0; j < 20; ++j)
                    rp = fmaf(sv[j], Wr[j * TD + col], rp);
                float w = (rp <= 2.0f)
                            ? 0.5f * (__cosf(PI_F * 0.5f * rp) + 1.f) : 0.f;
                g[gi] = w * ph[col];
            }
            accS[q] += g[1];
            float g2o = g[2] * orgn;
            accV[q][0] += g[0] * vp[c]            + g2o;
            accV[q][1] += g[0] * vp[DD + c]       + g2o;
            accV[q][2] += g[0] * vp[2 * DD + c]   + g2o;
        }
    }
    #pragma unroll
    for (int q = 0; q < 4; ++q) {
        int c = lane + 32 * q;
        atomicAdd(&ls[c], accS[q]);
        atomicAdd(&ls[128 + 0 * DD + c], accV[q][0]);
        atomicAdd(&ls[128 + 1 * DD + c], accV[q][1]);
        atomicAdd(&ls[128 + 2 * DD + c], accV[q][2]);
    }
    __syncthreads();
    for (int k = t; k < 512; k += 128) atomicAdd(&OSV[k], ls[k]);
}

// s += out_s (broadcast), v += out_v (broadcast); refresh bf16 mirrors.
__global__ void msg_apply(float* S32, __bf16* S16, float* V32, __bf16* V16,
                          const float* OSV, size_t n) {
    size_t idx = (size_t)blockIdx.x * 256 + threadIdx.x;
    if (idx >= n) return;
    int    c = (int)(idx & (DD - 1));
    size_t i = idx >> 7;
    float s = S32[idx] + OSV[c];
    S32[idx] = s; S16[idx] = f2b(s);
    #pragma unroll
    for (int d = 0; d < 3; ++d) {
        size_t vi = i * TD + (size_t)d * DD + c;
        float v = V32[vi] + OSV[128 + d * DD + c];
        V32[vi] = v; V16[vi] = f2b(v);
    }
}

// v += a0*u ; s += sdot*a1 + a2 ; refresh bf16 mirrors.
__global__ void upd_apply(float* S32, __bf16* S16, float* V32, __bf16* V16,
                          const float* A32, const float* U32,
                          const float* sdot, size_t n) {
    size_t idx = (size_t)blockIdx.x * 256 + threadIdx.x;
    if (idx >= n) return;
    int    c = (int)(idx & (DD - 1));
    size_t i = idx >> 7;
    const float* ar = A32 + i * TD;
    float a0 = ar[c], a1 = ar[DD + c], a2 = ar[2 * DD + c];
    float s = S32[idx] + sdot[0] * a1 + a2;
    S32[idx] = s; S16[idx] = f2b(s);
    #pragma unroll
    for (int d = 0; d < 3; ++d) {
        size_t vi = i * TD + (size_t)d * DD + c;
        float v = V32[vi] + a0 * U32[vi];
        V32[vi] = v; V16[vi] = f2b(v);
    }
}

// ---------------------------------------------------------------------------

extern "C" void kernel_launch(void* const* d_in, const int* in_sizes, int n_in,
                              void* d_out, int out_size, void* d_ws, size_t ws_size,
                              hipStream_t stream) {
    const int*   atomic = (const int*)  d_in[0];
    const float* r      = (const float*)d_in[1];
    const float* emb    = (const float*)d_in[2];
    const float* Wp1 = (const float*)d_in[3],  *bp1 = (const float*)d_in[4];
    const float* Wp2 = (const float*)d_in[5],  *bp2 = (const float*)d_in[6];
    const float* Wr  = (const float*)d_in[7],  *br  = (const float*)d_in[8];
    const float* Wa1 = (const float*)d_in[9],  *ba1 = (const float*)d_in[10];
    const float* Wa2 = (const float*)d_in[11], *ba2 = (const float*)d_in[12];
    const float* Wv  = (const float*)d_in[13], *bv  = (const float*)d_in[14];
    const float* Wu  = (const float*)d_in[15], *bu  = (const float*)d_in[16];
    const float* Wo1 = (const float*)d_in[17], *bo1 = (const float*)d_in[18];
    const float* Wo2 = (const float*)d_in[19], *bo2 = (const float*)d_in[20];

    const size_t N  = (size_t)in_sizes[0];   // atoms
    const size_t ND = N * DD;
    const size_t NT = N * TD;

    // ---- workspace carve-up (256B aligned) ----
    char* ws = (char*)d_ws;
    size_t off = 0;
    auto alloc = [&](size_t bytes) -> char* {
        char* p = ws + off;
        off += (bytes + 255) & ~(size_t)255;
        return p;
    };
    float*  S32  = (float*) alloc(ND * 4);
    __bf16* S16  = (__bf16*)alloc(ND * 2);
    float*  V32  = (float*) alloc(NT * 4);
    __bf16* V16  = (__bf16*)alloc(NT * 2);
    float*  PHI  = (float*) alloc(NT * 4);   // phi / a / final out
    float*  V2   = (float*) alloc(NT * 4);
    __bf16* V2_16= (__bf16*)alloc(NT * 2);
    float*  U    = (float*) alloc(NT * 4);
    __bf16* H16  = (__bf16*)alloc(ND * 2);
    float*  RN   = (float*) alloc(N * 4);
    __bf16* Wp1T = (__bf16*)alloc(128 * 128 * 2);
    __bf16* Wp2T = (__bf16*)alloc(384 * 128 * 2);
    __bf16* Wa1T = (__bf16*)alloc(128 * 128 * 2);   // lower half of Wa1
    __bf16* Wa2T = (__bf16*)alloc(384 * 128 * 2);
    __bf16* WvT  = (__bf16*)alloc(128 * 128 * 2);
    __bf16* WuT  = (__bf16*)alloc(128 * 128 * 2);
    __bf16* Wo1T = (__bf16*)alloc(128 * 128 * 2);
    __bf16* Wo2T = (__bf16*)alloc(128 * 128 * 2);
    float*  C1   = (float*) alloc(128 * 4);
    float*  OSV  = (float*) alloc(512 * 4);  // [0:128)=out_s, [128:512)=out_v
    float*  SCAL = (float*) alloc(4 * 4);    // [0]=1/frob, [1]=|v2|^2, [2]=sdot

    auto gemm = [&](const __bf16* A, const __bf16* BT, const float* bias,
                    const float* bias2, const float* scal, int scal_sqrt,
                    float* C32f, __bf16* C16, long M, int Nc, int act) {
        dim3 block(128);
        dim3 grid((unsigned)(M / 16));
        gemm_wmma<<<grid, block, 0, stream>>>(A, BT, bias, bias2, scal,
                                              scal_sqrt, C32f, C16, M, Nc, act);
    };
    auto rblocks = [](size_t n) -> unsigned {
        size_t b = (n + 255) / 256;
        return (unsigned)(b < 4096 ? b : 4096);
    };

    // ---- one-time prep ----
    wt_kernel<<<(128 * 128 + 255) / 256, 256, 0, stream>>>(Wp1, Wp1T, 128, 128, 0);
    wt_kernel<<<(384 * 128 + 255) / 256, 256, 0, stream>>>(Wp2, Wp2T, 128, 384, 0);
    wt_kernel<<<(128 * 128 + 255) / 256, 256, 0, stream>>>(Wa1, Wa1T, 128, 128, 128);
    wt_kernel<<<(384 * 128 + 255) / 256, 256, 0, stream>>>(Wa2, Wa2T, 128, 384, 0);
    wt_kernel<<<(128 * 128 + 255) / 256, 256, 0, stream>>>(Wv,  WvT,  128, 128, 0);
    wt_kernel<<<(128 * 128 + 255) / 256, 256, 0, stream>>>(Wu,  WuT,  128, 128, 0);
    wt_kernel<<<(128 * 128 + 255) / 256, 256, 0, stream>>>(Wo1, Wo1T, 128, 128, 0);
    wt_kernel<<<(128 * 128 + 255) / 256, 256, 0, stream>>>(Wo2, Wo2T, 128, 128, 0);
    c1_kernel<<<1, 128, 0, stream>>>(Wa1, C1);

    zero_kernel<<<4096, 256, 0, stream>>>(V32, NT);
    zero_kernel<<<4096, 256, 0, stream>>>((float*)V16, NT / 2);
    embed_kernel<<<(unsigned)((ND + 255) / 256), 256, 0, stream>>>(atomic, emb, S32, S16, ND);
    zero_kernel<<<1, 32, 0, stream>>>(SCAL, 4);
    rn_kernel<<<(unsigned)((N + 255) / 256), 256, 0, stream>>>(r, RN, &SCAL[0], (int)N);
    k_rsqrt<<<1, 1, 0, stream>>>(&SCAL[0]);           // SCAL[0] = 1/||r||_F

    // ---- 3 interaction blocks ----
    for (int it = 0; it < 3; ++it) {
        zero_kernel<<<1, 128, 0, stream>>>(OSV, 512);
        zero_kernel<<<1, 32, 0, stream>>>(&SCAL[1], 2);

        // message: phi = silu(s@Wp1+bp1)@Wp2+bp2
        gemm(S16, Wp1T, bp1, nullptr, nullptr, 0, nullptr, H16, (long)N, 128, 1);
        gemm(H16, Wp2T, bp2, nullptr, nullptr, 0, PHI, nullptr, (long)N, 384, 0);
        gate_reduce_kernel<<<1024, 128, 0, stream>>>(PHI, V32, RN, &SCAL[0], Wr, br, OSV, (int)N);
        msg_apply<<<(unsigned)((ND + 255) / 256), 256, 0, stream>>>(S32, S16, V32, V16, OSV, ND);

        // update: v2 = v@Wv+bv ; u = v2@Wu+bu
        gemm(V16, WvT, bv, nullptr, nullptr, 0, V2, V2_16, (long)(3 * N), 128, 0);
        reduce_sq<<<rblocks(NT), 256, 0, stream>>>(V2, NT, &SCAL[1]);     // ||v2||^2
        gemm(V2_16, WuT, bu, nullptr, nullptr, 0, U, nullptr, (long)(3 * N), 128, 0);
        reduce_dot<<<rblocks(NT), 256, 0, stream>>>(V2, U, NT, &SCAL[2]); // sdot

        // a = silu([vnorm,s]@Wa1+ba1)@Wa2+ba2, with vnorm folded into bias
        gemm(S16, Wa1T, ba1, C1, &SCAL[1], /*sqrt*/1, nullptr, H16, (long)N, 128, 1);
        gemm(H16, Wa2T, ba2, nullptr, nullptr, 0, PHI, nullptr, (long)N, 384, 0);
        upd_apply<<<(unsigned)((ND + 255) / 256), 256, 0, stream>>>(S32, S16, V32, V16,
                                                                    PHI, U, &SCAL[2], ND);
    }

    // ---- output head: sum(silu(s@Wo1+bo1)@Wo2+bo2) ----
    gemm(S16, Wo1T, bo1, nullptr, nullptr, 0, nullptr, H16, (long)N, 128, 1);
    gemm(H16, Wo2T, bo2, nullptr, nullptr, 0, PHI, nullptr, (long)N, 128, 0);
    zero_kernel<<<1, 32, 0, stream>>>((float*)d_out, (size_t)out_size);
    reduce_sum<<<rblocks(ND), 256, 0, stream>>>(PHI, ND, (float*)d_out);
}
